// MessagePassingGNN_22419729285671
// MI455X (gfx1250) — compile-verified, matrix operand order checked by reading, not run on hardware
//
#include <hip/hip_runtime.h>
#include <hip/hip_bf16.h>

typedef __attribute__((ext_vector_type(16))) _Float16 v16h;
typedef __attribute__((ext_vector_type(8)))  _Float16 v8h;
typedef __attribute__((ext_vector_type(8)))  float    v8f;

#define H   128
#define DH  256
#define ROWPA 264   // padded row length (halfs) for 256-wide LDS tiles (33*16B -> conflict free)
#define ROWPG 136   // padded row length (halfs) for 128-wide LDS tiles

union F8 { v8f v; float f[8]; };

// Branch-free fast tanh: clamp + v_exp_f32 + v_rcp_f32. |err| < 1e-6 rel inside
// clamp, exact saturation outside -- far below the f16 WMMA quantization noise.
__device__ __forceinline__ float tanh_fast(float x) {
  x = fminf(9.0f, fmaxf(-9.0f, x));
  float t = __expf(2.0f * x);
  return __fdividef(t - 1.0f, t + 1.0f);
}

__device__ __forceinline__ float sigmoid_fast(float x) {
  return __fdividef(1.0f, 1.0f + __expf(-x));
}

// Build 16x32 f16 A-fragments (per CDNA5 ISA layout) from a row-major row of halfs.
// lanes 0-15: M=lane, K = {kt*32+0..7, kt*32+16..23}; lanes 16-31: M=lane-16, K = {+8..15, +24..31}
template<int KT>
__device__ __forceinline__ void load_a_frags(v16h* a, const _Float16* rowbase, int hi) {
#pragma unroll
  for (int kt = 0; kt < KT; ++kt) {
    const v8h* p = (const v8h*)(rowbase + kt * 32);
    v8h lo = p[hi ? 1 : 0];
    v8h hh = p[hi ? 3 : 2];
    a[kt] = __builtin_shufflevector(lo, hh, 0,1,2,3,4,5,6,7,8,9,10,11,12,13,14,15);
  }
}

// One 16x16 output tile: accumulate KT WMMAs over pre-swizzled B fragments.
template<int KT>
__device__ __forceinline__ v8f wmma_ntile(const v16h* a, const _Float16* __restrict__ wf,
                                          int nt, int NT, int lane) {
  v8f acc = {0.f,0.f,0.f,0.f,0.f,0.f,0.f,0.f};
  __builtin_prefetch(wf + (((long)(nt + 1)) * 32) * 16);   // global_prefetch_b8 of next B group
#pragma unroll
  for (int kt = 0; kt < KT; ++kt) {
    v16h b = *(const v16h*)(wf + (((long)kt * NT + nt) * 32 + lane) * 16);
    acc = __builtin_amdgcn_wmma_f32_16x16x32_f16(false, a[kt], false, b,
                                                 (short)0, acc, false, false);
  }
  return acc;
}

// ---------------- weight pre-swizzle: f32 [K,N] (or [N,K] if transposed) -> f16 B fragments
__global__ void swz_kernel(const float* __restrict__ W, _Float16* __restrict__ out,
                           int K, int N, int transposed) {
  long idx = (long)blockIdx.x * blockDim.x + threadIdx.x;
  long total = (long)K * N;
  if (idx >= total) return;
  int e    = (int)(idx & 15);
  int lane = (int)((idx >> 4) & 31);
  long frag = idx >> 9;
  int NT = N >> 4;
  int kt = (int)(frag / NT);
  int nt = (int)(frag % NT);
  int n = nt * 16 + (lane & 15);
  int koff = (lane < 16) ? ((e < 8) ? e : e + 8) : ((e < 8) ? e + 8 : e + 16);
  int k = kt * 32 + koff;
  float v = transposed ? W[(long)n * K + k] : W[(long)k * N + n];
  out[idx] = (_Float16)v;
}

// ---------------- per-dst degree counts (self loops appended) ----------------
__global__ void count_kernel(const int* __restrict__ dstp, float* __restrict__ cnt,
                             int nE, int nNodes) {
  int e = blockIdx.x * blockDim.x + threadIdx.x;
  int ET = nE + nNodes;
  if (e >= ET) return;
  int d = (e < nE) ? dstp[e] : (e - nE);
  atomicAdd(&cnt[d], 1.0f);
}

// ---------------- encoder: h = tanh(x @ enc_W + enc_b), also f16 copy ----------------
__global__ void __launch_bounds__(128)
enc_kernel(const float* __restrict__ x, const float* __restrict__ W,
           const float* __restrict__ b, float* __restrict__ h,
           _Float16* __restrict__ hf, int nNodes) {
  __shared__ float xs[64];
  int n = blockIdx.x;
  int t = threadIdx.x;
  if (t < 64) xs[t] = x[(long)n * 64 + t];
  __syncthreads();
  float acc = b[t];
#pragma unroll
  for (int k = 0; k < 64; ++k) acc += xs[k] * W[k * H + t];
  float r = tanh_fast(acc);
  h[(long)n * H + t] = r;
  hf[(long)n * H + t] = (_Float16)r;
}

// ---------------- message MLP over edges + atomic scatter aggregation ----------------
__global__ void __launch_bounds__(64)
msg_kernel(const _Float16* __restrict__ hf,
           const int* __restrict__ srcp, const int* __restrict__ dstp,
           const _Float16* __restrict__ w0, const _Float16* __restrict__ w1,
           const _Float16* __restrict__ w2,
           const float* __restrict__ b0, const float* __restrict__ b1,
           const float* __restrict__ b2,
           float* __restrict__ aggr, int nE, int nNodes) {
  __shared__ _Float16 sA[2][16 * ROWPA];
  __shared__ _Float16 sB[2][16 * ROWPA];
  __shared__ int sDst[2][16];
  const int lane = threadIdx.x & 31;
  const int wave = threadIdx.x >> 5;
  const int l15 = lane & 15;
  const int hi  = lane >> 4;
  const int ET  = nE + nNodes;
  const int tile = blockIdx.x * 2 + wave;
  const int e0 = tile * 16;
  if (e0 >= ET) return;

  // gather [h[dst] || h[src]] (f16) into LDS; lane pair (m, part) copies one 128-half row
  {
    int m = lane >> 1, part = lane & 1;
    int e = e0 + m;
    int s = 0, d = 0;
    if (e < nE)      { s = srcp[e]; d = dstp[e]; }
    else if (e < ET) { s = d = e - nE; }
    if (part == 0) sDst[wave][m] = d;
    const float4* sp = (const float4*)(hf + (long)(part ? s : d) * H);
    float4* dp = (float4*)(&sA[wave][m * ROWPA + part * H]);
#pragma unroll
    for (int i = 0; i < 16; ++i) dp[i] = sp[i];
  }

  v16h a[8];
  F8 u;

  // GEMM0: [16 x 2H] x [2H x DH] + b0, tanh -> sB
  asm volatile("s_wait_dscnt 0x0" ::: "memory");
  load_a_frags<8>(a, &sA[wave][l15 * ROWPA], hi);
  for (int nt = 0; nt < 16; ++nt) {
    u.v = wmma_ntile<8>(a, w0, nt, 16, lane);
    int col = nt * 16 + l15;
    float bb = b0[col];
#pragma unroll
    for (int v = 0; v < 8; ++v)
      sB[wave][(hi * 8 + v) * ROWPA + col] = (_Float16)tanh_fast(u.f[v] + bb);
  }

  // GEMM1: [16 x DH] x [DH x DH] + b1, tanh -> sA (input no longer needed)
  asm volatile("s_wait_dscnt 0x0" ::: "memory");
  load_a_frags<8>(a, &sB[wave][l15 * ROWPA], hi);
  for (int nt = 0; nt < 16; ++nt) {
    u.v = wmma_ntile<8>(a, w1, nt, 16, lane);
    int col = nt * 16 + l15;
    float bb = b1[col];
#pragma unroll
    for (int v = 0; v < 8; ++v)
      sA[wave][(hi * 8 + v) * ROWPA + col] = (_Float16)tanh_fast(u.f[v] + bb);
  }

  // GEMM2: [16 x DH] x [DH x H] + b2, scatter-add into aggr[dst]
  asm volatile("s_wait_dscnt 0x0" ::: "memory");
  load_a_frags<8>(a, &sA[wave][l15 * ROWPA], hi);
  for (int nt = 0; nt < 8; ++nt) {
    u.v = wmma_ntile<8>(a, w2, nt, 8, lane);
    int col = nt * 16 + l15;
    float bb = b2[col];
#pragma unroll
    for (int v = 0; v < 8; ++v) {
      int m = hi * 8 + v;
      if (e0 + m < ET)
        atomicAdd(&aggr[(long)sDst[wave][m] * H + col], u.f[v] + bb);
    }
  }
}

// ---------------- GRU cell: mean-normalize aggr, two WMMA GEMMs, fused gates ----------------
__global__ void __launch_bounds__(128)
gru_kernel(float* __restrict__ h, _Float16* __restrict__ hf,
           const float* __restrict__ aggr, const float* __restrict__ cnt,
           const _Float16* __restrict__ wih, const _Float16* __restrict__ whh,
           const float* __restrict__ bih, const float* __restrict__ bhh,
           int nNodes) {
  __shared__ _Float16 sG[4][16 * ROWPG];
  const int lane = threadIdx.x & 31;
  const int wave = threadIdx.x >> 5;
  const int l15 = lane & 15, hi = lane >> 4;
  const int tile = blockIdx.x * 4 + wave;
  const int node0 = tile * 16;
  if (node0 >= nNodes) return;

  { // aggr / count -> f16 LDS tile
    int m = lane >> 1, part = lane & 1;
    int n = node0 + m; if (n >= nNodes) n = nNodes - 1;
    float inv = __fdividef(1.0f, cnt[n]);
    const float* ar = aggr + (long)n * H + part * 64;
    _Float16* dr = &sG[wave][m * ROWPG + part * 64];
    for (int i = 0; i < 64; ++i) dr[i] = (_Float16)(ar[i] * inv);
  }

  v16h ag[4], hfr[4];
  asm volatile("s_wait_dscnt 0x0" ::: "memory");
  load_a_frags<4>(ag, &sG[wave][l15 * ROWPG], hi);
  {
    int n = node0 + l15; if (n >= nNodes) n = nNodes - 1;
    load_a_frags<4>(hfr, hf + (long)n * H, hi);  // global b128 loads
  }

  for (int j = 0; j < 8; ++j) {   // 8 column tiles of H; gate chunks at j, j+8, j+16
    F8 ir, iz, in_, hr, hz, hn;
    ir.v  = wmma_ntile<4>(ag,  wih, j,      24, lane);
    iz.v  = wmma_ntile<4>(ag,  wih, j + 8,  24, lane);
    in_.v = wmma_ntile<4>(ag,  wih, j + 16, 24, lane);
    hr.v  = wmma_ntile<4>(hfr, whh, j,      24, lane);
    hz.v  = wmma_ntile<4>(hfr, whh, j + 8,  24, lane);
    hn.v  = wmma_ntile<4>(hfr, whh, j + 16, 24, lane);
    int col = j * 16 + l15;
    float bir = bih[col], biz = bih[col + H], bin = bih[col + 2 * H];
    float bhr = bhh[col], bhz = bhh[col + H], bhn = bhh[col + 2 * H];
#pragma unroll
    for (int v = 0; v < 8; ++v) {
      int n = node0 + hi * 8 + v;
      if (n < nNodes) {
        long idx = (long)n * H + col;
        float ho = h[idx];
        float r = sigmoid_fast(ir.f[v] + bir + hr.f[v] + bhr);
        float z = sigmoid_fast(iz.f[v] + biz + hz.f[v] + bhz);
        float nn = tanh_fast(in_.f[v] + bin + r * (hn.f[v] + bhn));
        float hnew = (1.0f - z) * nn + z * ho;
        h[idx] = hnew;
        hf[idx] = (_Float16)hnew;
      }
    }
  }
}

// ---------------- decoder: tanh(128->256) tanh(256->256) -> 256->1 ----------------
__global__ void __launch_bounds__(64)
dec_kernel(const _Float16* __restrict__ hf,
           const _Float16* __restrict__ w0, const _Float16* __restrict__ w1,
           const float* __restrict__ b0, const float* __restrict__ b1,
           const float* __restrict__ w2, const float* __restrict__ b2,
           float* __restrict__ y, int nNodes) {
  __shared__ _Float16 sB[2][16 * ROWPA];
  __shared__ _Float16 sC[2][16 * ROWPA];
  const int lane = threadIdx.x & 31;
  const int wave = threadIdx.x >> 5;
  const int l15 = lane & 15, hi = lane >> 4;
  const int tile = blockIdx.x * 2 + wave;
  const int node0 = tile * 16;
  if (node0 >= nNodes) return;

  F8 u;
  { // GEMM0 from global f16 h rows
    v16h a[4];
    int n = node0 + l15; if (n >= nNodes) n = nNodes - 1;
    load_a_frags<4>(a, hf + (long)n * H, hi);
    for (int nt = 0; nt < 16; ++nt) {
      u.v = wmma_ntile<4>(a, w0, nt, 16, lane);
      int col = nt * 16 + l15;
      float bb = b0[col];
#pragma unroll
      for (int v = 0; v < 8; ++v)
        sB[wave][(hi * 8 + v) * ROWPA + col] = (_Float16)tanh_fast(u.f[v] + bb);
    }
  }
  { // GEMM1
    v16h a[8];
    asm volatile("s_wait_dscnt 0x0" ::: "memory");
    load_a_frags<8>(a, &sB[wave][l15 * ROWPA], hi);
    for (int nt = 0; nt < 16; ++nt) {
      u.v = wmma_ntile<8>(a, w1, nt, 16, lane);
      int col = nt * 16 + l15;
      float bb = b1[col];
#pragma unroll
      for (int v = 0; v < 8; ++v)
        sC[wave][(hi * 8 + v) * ROWPA + col] = (_Float16)tanh_fast(u.f[v] + bb);
    }
  }
  asm volatile("s_wait_dscnt 0x0" ::: "memory");
  { // final 256 -> 1 dot, pair reduction across (part) lanes
    int m = lane >> 1, part = lane & 1;
    int n = node0 + m;
    const _Float16* row = &sC[wave][m * ROWPA + part * 128];
    float s = 0.f;
    for (int i = 0; i < 128; ++i) s += (float)row[i] * w2[part * 128 + i];
    s += __shfl_xor(s, 1);
    if (part == 0 && n < nNodes) y[n] = s + b2[0];
  }
}

extern "C" void kernel_launch(void* const* d_in, const int* in_sizes, int n_in,
                              void* d_out, int out_size, void* d_ws, size_t ws_size,
                              hipStream_t stream) {
  (void)n_in; (void)out_size; (void)ws_size;
  const float* x    = (const float*)d_in[0];
  const int*   ei   = (const int*)d_in[1];
  const float* encW = (const float*)d_in[2];
  const float* encb = (const float*)d_in[3];
  const float* mW0  = (const float*)d_in[4];
  const float* mb0  = (const float*)d_in[5];
  const float* mW1  = (const float*)d_in[6];
  const float* mb1  = (const float*)d_in[7];
  const float* mW2  = (const float*)d_in[8];
  const float* mb2  = (const float*)d_in[9];
  const float* gWih = (const float*)d_in[10];
  const float* gWhh = (const float*)d_in[11];
  const float* gbih = (const float*)d_in[12];
  const float* gbhh = (const float*)d_in[13];
  const float* dW0  = (const float*)d_in[14];
  const float* db0  = (const float*)d_in[15];
  const float* dW1  = (const float*)d_in[16];
  const float* db1  = (const float*)d_in[17];
  const float* dW2  = (const float*)d_in[18];
  const float* db2  = (const float*)d_in[19];

  const int nNodes = in_sizes[0] / 64;
  const int nE     = in_sizes[1] / 2;
  const int ET     = nE + nNodes;
  const int* srcp = ei;
  const int* dstp = ei + nE;

  // ---- workspace carve-up ----
  char* ws = (char*)d_ws;
  size_t off = 0;
  auto take = [&](size_t bytes) -> char* {
    char* p = ws + off;
    off = (off + bytes + 255) & ~(size_t)255;
    return p;
  };
  float*     h    = (float*)take((size_t)nNodes * H * 4);
  _Float16*  hfp  = (_Float16*)take((size_t)nNodes * H * 2);
  float*     aggr = (float*)take((size_t)nNodes * H * 4);
  float*     cnt  = (float*)take((size_t)nNodes * 4);

  const size_t SZ_M0 = (size_t)2 * H * DH;   // 65536
  const size_t SZ_M1 = (size_t)DH * DH;      // 65536
  const size_t SZ_M2 = (size_t)DH * H;       // 32768
  const size_t SZ_G  = (size_t)H * 3 * H;    // 49152
  const size_t SZ_D0 = (size_t)H * DH;       // 32768
  const size_t SZ_D1 = (size_t)DH * DH;      // 65536
  size_t totw = 3 * (SZ_M0 + SZ_M1 + SZ_M2) + 3 * 2 * SZ_G + SZ_D0 + SZ_D1;
  _Float16* wbuf = (_Float16*)take(totw * 2);

  _Float16 *wm0[3], *wm1[3], *wm2[3], *wgi[3], *wgh[3], *wd0, *wd1;
  size_t wo = 0;
  for (int l = 0; l < 3; ++l) {
    wm0[l] = wbuf + wo; wo += SZ_M0;
    wm1[l] = wbuf + wo; wo += SZ_M1;
    wm2[l] = wbuf + wo; wo += SZ_M2;
  }
  for (int l = 0; l < 3; ++l) {
    wgi[l] = wbuf + wo; wo += SZ_G;
    wgh[l] = wbuf + wo; wo += SZ_G;
  }
  wd0 = wbuf + wo; wo += SZ_D0;
  wd1 = wbuf + wo; wo += SZ_D1;

  // ---- degree counts ----
  hipMemsetAsync(cnt, 0, (size_t)nNodes * 4, stream);
  count_kernel<<<(ET + 255) / 256, 256, 0, stream>>>(dstp, cnt, nE, nNodes);

  // ---- encoder ----
  enc_kernel<<<nNodes, 128, 0, stream>>>(x, encW, encb, h, hfp, nNodes);

  // ---- weight pre-swizzle into WMMA B-fragment layout (f16) ----
  auto swz = [&](const float* W, _Float16* out, int K, int N, int tr) {
    long tot = (long)K * N;
    swz_kernel<<<(int)((tot + 255) / 256), 256, 0, stream>>>(W, out, K, N, tr);
  };
  for (int l = 0; l < 3; ++l) {
    swz(mW0 + (size_t)l * SZ_M0, wm0[l], 2 * H, DH, 0);
    swz(mW1 + (size_t)l * SZ_M1, wm1[l], DH, DH, 0);
    swz(mW2 + (size_t)l * SZ_M2, wm2[l], DH, H, 0);
    swz(gWih + (size_t)l * SZ_G, wgi[l], H, 3 * H, 1);  // torch [3H,H] -> W^T
    swz(gWhh + (size_t)l * SZ_G, wgh[l], H, 3 * H, 1);
  }
  swz(dW0, wd0, H, DH, 0);
  swz(dW1, wd1, DH, DH, 0);

  // ---- propagation steps ----
  const int etiles = (ET + 15) / 16;
  const int ntiles = (nNodes + 15) / 16;
  for (int l = 0; l < 3; ++l) {
    hipMemsetAsync(aggr, 0, (size_t)nNodes * H * 4, stream);
    msg_kernel<<<(etiles + 1) / 2, 64, 0, stream>>>(
        hfp, srcp, dstp, wm0[l], wm1[l], wm2[l],
        mb0 + (size_t)l * DH, mb1 + (size_t)l * DH, mb2 + (size_t)l * H,
        aggr, nE, nNodes);
    gru_kernel<<<(ntiles + 3) / 4, 128, 0, stream>>>(
        h, hfp, aggr, cnt, wgi[l], wgh[l],
        gbih + (size_t)l * 3 * H, gbhh + (size_t)l * 3 * H, nNodes);
  }

  // ---- decoder ----
  dec_kernel<<<(ntiles + 1) / 2, 64, 0, stream>>>(
      hfp, wd0, wd1, db0, db1, dW2, db2, (float*)d_out, nNodes);
}
